// AR_Decoder_43645457662274
// MI455X (gfx1250) — matrix operand order repaired and checked by reading, not compiled, not run on hardware
//
#include <hip/hip_runtime.h>
#include <hip/hip_bf16.h>

typedef __bf16 bf16_t;
typedef __attribute__((ext_vector_type(16))) __bf16 v16bf;
typedef __attribute__((ext_vector_type(8)))  __bf16 v8bf;
typedef __attribute__((ext_vector_type(4)))  __bf16 v4bf;
typedef __attribute__((ext_vector_type(8)))  float  v8f;

#define HID    1024
#define BROWS  4096
#define KPSZ   256

#define BM 128        // batch rows per workgroup
#define BN 32         // hidden cols per workgroup (per gate)
#define KC 64         // K chunk = 2 WMMA k-steps
#define LDX 72        // padded LDS stride (elements): 144 B, 16B-aligned rows

// ---------------------------------------------------------------------------
// fp32 -> bf16 conversion, float4-vectorized grid-stride
// ---------------------------------------------------------------------------
__global__ void cvt_f32_to_bf16(const float4* __restrict__ in,
                                v4bf* __restrict__ out, int n4) {
    int i = blockIdx.x * blockDim.x + threadIdx.x;
    int s = gridDim.x * blockDim.x;
    for (; i < n4; i += s) {
        float4 v = in[i];
        v4bf o;
        o[0] = (__bf16)v.x; o[1] = (__bf16)v.y;
        o[2] = (__bf16)v.z; o[3] = (__bf16)v.w;
        out[i] = o;
    }
}

// ---------------------------------------------------------------------------
// CDNA5 async global->LDS copy (ASYNCcnt path, no VGPR staging).
// ISA 15.18.3: data at global vaddr lands in LDS at vdst; the immediate offset adds to both.
// ---------------------------------------------------------------------------
__device__ __forceinline__ void async_copy_64B(unsigned lds_off,
                                               const bf16_t* gptr) {
    unsigned long long ga = (unsigned long long)(uintptr_t)gptr;
    asm volatile(
        "global_load_async_to_lds_b128 %0, %1, off\n\t"
        "global_load_async_to_lds_b128 %0, %1, off offset:16\n\t"
        "global_load_async_to_lds_b128 %0, %1, off offset:32\n\t"
        "global_load_async_to_lds_b128 %0, %1, off offset:48"
        :: "v"(lds_off), "v"(ga) : "memory");
}
__device__ __forceinline__ void wait_async_zero() {
    asm volatile("s_wait_asynccnt 0x0" ::: "memory");
}

// ---------------------------------------------------------------------------
// fast activations
// ---------------------------------------------------------------------------
__device__ __forceinline__ float fast_sigmoid(float x) {
    return __builtin_amdgcn_rcpf(1.0f + __expf(-x));
}
__device__ __forceinline__ float fast_tanh(float x) {
    float e = __expf(-2.0f * x);
    return (1.0f - e) * __builtin_amdgcn_rcpf(1.0f + e);
}

// ---------------------------------------------------------------------------
// Fragment builders (ISA §7.12.2 16-bit layouts, wave32)
//  A 16x32: lanes0-15 row=lr elems {K0..7,K16..23}; lanes16-31 {K8..15,K24..31}
//  B 32x16: lanes0-15 col=lr elems {K0..7,K16..23}; lanes16-31 {K8..15,K24..31}
// ---------------------------------------------------------------------------
__device__ __forceinline__ v16bf load_a_frag(const bf16_t* rowp, int lhi) {
    const bf16_t* ap = rowp + lhi * 8;
    v8bf lo = *(const v8bf*)(ap);
    v8bf hi = *(const v8bf*)(ap + 16);
    v16bf a;
#pragma unroll
    for (int i = 0; i < 8; ++i) { a[i] = lo[i]; a[i + 8] = hi[i]; }
    return a;
}
__device__ __forceinline__ v16bf load_b_frag(const bf16_t* colrowp, int lhi) {
    const bf16_t* bp = colrowp + lhi * 8;
    v8bf lo = *(const v8bf*)(bp);
    v8bf hi = *(const v8bf*)(bp + 16);
    v16bf b;
#pragma unroll
    for (int i = 0; i < 8; ++i) { b[i] = lo[i]; b[i + 8] = hi[i]; }
    return b;
}

// ---------------------------------------------------------------------------
// Fused LSTM cell, double-buffered async pipeline.
// Tile: BM x BN x 4 gates per block; 8 waves, wave w owns rows [w*16, w*16+16).
// ---------------------------------------------------------------------------
struct __align__(16) SmemCell {
    bf16_t x[2][BM][LDX];       // activation tiles (double buffered)
    bf16_t w[2][4][BN][LDX];    // weight rows per gate
};

template <int KIN>
__global__ __launch_bounds__(256)
void lstm_cell_kernel(const bf16_t* __restrict__ xbf,   // [B, KIN]
                      const bf16_t* __restrict__ hbf,   // [B, HID]
                      const bf16_t* __restrict__ wih,   // [4H, KIN]
                      const bf16_t* __restrict__ whh,   // [4H, HID]
                      const float*  __restrict__ bih,   // [4H]
                      const float*  __restrict__ bhh,   // [4H]
                      const float*  __restrict__ cin,   // [B, H]
                      float* __restrict__ hout,
                      float* __restrict__ hout2,        // optional dup
                      float* __restrict__ cout) {
    __shared__ SmemCell sm;

    const int tid  = threadIdx.x;
    const int lane = tid & 31;
    const int wave = tid >> 5;          // 0..7
    const int lr   = lane & 15;
    const int lhi  = lane >> 4;
    const int n0   = blockIdx.x * BN;   // hidden col block
    const int m0   = blockIdx.y * BM;   // batch row block

    v8f acc[4][2];
#pragma unroll
    for (int g = 0; g < 4; ++g)
#pragma unroll
        for (int ct = 0; ct < 2; ++ct)
#pragma unroll
            for (int r = 0; r < 8; ++r) acc[g][ct][r] = 0.0f;

    // per-thread staging coords: 32 bf16 (64 B) each for x and w per chunk
    const int xrow = tid >> 1;          // 0..127
    const int xko  = (tid & 1) * 32;
    const int wg   = tid >> 6;          // gate 0..3
    const int wrow = (tid & 63) >> 1;   // 0..31
    const int wko  = (tid & 1) * 32;

    const int NCH0 = KIN / KC;          // chunks in phase 0
    const int NCH  = NCH0 + HID / KC;   // total chunks

    auto stage = [&](int buf, int chunk) {
        const bf16_t* act;
        const bf16_t* wgt;
        int K, k0;
        if (chunk < NCH0) { act = xbf; wgt = wih; K = KIN; k0 = chunk * KC; }
        else              { act = hbf; wgt = whh; K = HID; k0 = (chunk - NCH0) * KC; }
        async_copy_64B((unsigned)(uintptr_t)&sm.x[buf][xrow][xko],
                       act + (size_t)(m0 + xrow) * K + k0 + xko);
        async_copy_64B((unsigned)(uintptr_t)&sm.w[buf][wg][wrow][wko],
                       wgt + (size_t)(wg * HID + n0 + wrow) * K + k0 + wko);
    };

    auto compute = [&](int buf) {
#pragma unroll
        for (int s = 0; s < 2; ++s) {           // two WMMA k-steps per chunk
            v16bf a = load_a_frag(&sm.x[buf][wave * 16 + lr][s * 32], lhi);
            v16bf b[4][2];
#pragma unroll
            for (int g = 0; g < 4; ++g)
#pragma unroll
                for (int ct = 0; ct < 2; ++ct)
                    b[g][ct] = load_b_frag(&sm.w[buf][g][ct * 16 + lr][s * 32], lhi);
#pragma unroll
            for (int g = 0; g < 4; ++g)
#pragma unroll
                for (int ct = 0; ct < 2; ++ct)
                    acc[g][ct] = __builtin_amdgcn_wmma_f32_16x16x32_bf16(
                        false, a, false, b[g][ct], (short)0, acc[g][ct],
                        false, false);
        }
    };

    // software pipeline: prefetch chunk c+1 into buf^1 while computing buf
    stage(0, 0);
    wait_async_zero();
    __syncthreads();
    int buf = 0;
#pragma unroll 1
    for (int c = 0; c < NCH; ++c) {
        if (c + 1 < NCH) stage(buf ^ 1, c + 1);
        compute(buf);
        wait_async_zero();
        __syncthreads();
        buf ^= 1;
    }

    // ---- fused LSTM epilogue -------------------------------------------
    float bias[4][2];
#pragma unroll
    for (int g = 0; g < 4; ++g)
#pragma unroll
        for (int ct = 0; ct < 2; ++ct) {
            int n = g * HID + n0 + ct * 16 + lr;
            bias[g][ct] = bih[n] + bhh[n];
        }

#pragma unroll
    for (int ct = 0; ct < 2; ++ct) {
        const int n = n0 + ct * 16 + lr;
#pragma unroll
        for (int r = 0; r < 8; ++r) {
            const int m = m0 + wave * 16 + lhi * 8 + r;
            const size_t idx = (size_t)m * HID + n;
            float iv = fast_sigmoid(acc[0][ct][r] + bias[0][ct]);
            float fv = fast_sigmoid(acc[1][ct][r] + bias[1][ct]);
            float gv = fast_tanh   (acc[2][ct][r] + bias[2][ct]);
            float ov = fast_sigmoid(acc[3][ct][r] + bias[3][ct]);
            float cn = fv * cin[idx] + iv * gv;
            float hn = ov * fast_tanh(cn);
            cout[idx] = cn;
            hout[idx] = hn;
            if (hout2) hout2[idx] = hn;
        }
    }
}

// ---------------------------------------------------------------------------
// Embedding GEMM: frame = kps @ W_emb^T + b_emb, bf16 out (input to cell 1)
// Small (0.3% of FLOPs) — keep simple synchronous staging, KC=32.
// ---------------------------------------------------------------------------
#define EKC 32
#define ELD (EKC + 8)
struct __align__(16) SmemEmb {
    bf16_t x[BM][ELD];
    bf16_t w[BN][ELD];
};

__global__ __launch_bounds__(256)
void emb_kernel(const bf16_t* __restrict__ kpsbf,   // [B, KPSZ]
                const bf16_t* __restrict__ wembbf,  // [KPSZ, KPSZ]
                const float*  __restrict__ bemb,    // [KPSZ]
                bf16_t* __restrict__ outbf) {       // [B, KPSZ]
    __shared__ SmemEmb sm;

    const int tid  = threadIdx.x;
    const int lane = tid & 31;
    const int wave = tid >> 5;
    const int lr   = lane & 15;
    const int lhi  = lane >> 4;
    const int n0   = blockIdx.x * BN;
    const int m0   = blockIdx.y * BM;

    v8f acc[2];
#pragma unroll
    for (int ct = 0; ct < 2; ++ct)
#pragma unroll
        for (int r = 0; r < 8; ++r) acc[ct][r] = 0.0f;

    const int xrow = tid >> 1;
    const int xko  = (tid & 1) * 16;

#pragma unroll 1
    for (int k0 = 0; k0 < KPSZ; k0 += EKC) {
        __syncthreads();
        {
            const bf16_t* src = kpsbf + (size_t)(m0 + xrow) * KPSZ + k0 + xko;
            uint4 d0 = *(const uint4*)(src);
            uint4 d1 = *(const uint4*)(src + 8);
            *(uint4*)(&sm.x[xrow][xko])     = d0;
            *(uint4*)(&sm.x[xrow][xko + 8]) = d1;
        }
        if (tid < 64) {   // 32 rows x 32 k
            int row = tid >> 1;
            int ko  = (tid & 1) * 16;
            const bf16_t* src = wembbf + (size_t)(n0 + row) * KPSZ + k0 + ko;
            uint4 d0 = *(const uint4*)(src);
            uint4 d1 = *(const uint4*)(src + 8);
            *(uint4*)(&sm.w[row][ko])     = d0;
            *(uint4*)(&sm.w[row][ko + 8]) = d1;
        }
        __syncthreads();

        v16bf a = load_a_frag(&sm.x[wave * 16 + lr][0], lhi);
        v16bf b0 = load_b_frag(&sm.w[lr][0], lhi);
        v16bf b1 = load_b_frag(&sm.w[16 + lr][0], lhi);
        acc[0] = __builtin_amdgcn_wmma_f32_16x16x32_bf16(
            false, a, false, b0, (short)0, acc[0], false, false);
        acc[1] = __builtin_amdgcn_wmma_f32_16x16x32_bf16(
            false, a, false, b1, (short)0, acc[1], false, false);
    }

#pragma unroll
    for (int ct = 0; ct < 2; ++ct) {
        const int n = n0 + ct * 16 + lr;
        const float bb = bemb[n];
#pragma unroll
        for (int r = 0; r < 8; ++r) {
            const int m = m0 + wave * 16 + lhi * 8 + r;
            outbf[(size_t)m * KPSZ + n] = (__bf16)(acc[ct][r] + bb);
        }
    }
}

// ---------------------------------------------------------------------------
// Host launcher
// ---------------------------------------------------------------------------
extern "C" void kernel_launch(void* const* d_in, const int* in_sizes, int n_in,
                              void* d_out, int out_size, void* d_ws, size_t ws_size,
                              hipStream_t stream) {
    (void)in_sizes; (void)n_in; (void)out_size; (void)ws_size;

    const float* kps   = (const float*)d_in[0];
    const float* h0    = (const float*)d_in[1];
    const float* h1    = (const float*)d_in[2];
    const float* h2    = (const float*)d_in[3];
    const float* c0    = (const float*)d_in[4];
    const float* c1    = (const float*)d_in[5];
    const float* c2    = (const float*)d_in[6];
    const float* W_emb = (const float*)d_in[7];
    const float* b_emb = (const float*)d_in[8];
    const float* w_ih1 = (const float*)d_in[9];
    const float* w_hh1 = (const float*)d_in[10];
    const float* b_ih1 = (const float*)d_in[11];
    const float* b_hh1 = (const float*)d_in[12];
    const float* w_ih2 = (const float*)d_in[13];
    const float* w_hh2 = (const float*)d_in[14];
    const float* b_ih2 = (const float*)d_in[15];
    const float* b_hh2 = (const float*)d_in[16];
    const float* w_ih3 = (const float*)d_in[17];
    const float* w_hh3 = (const float*)d_in[18];
    const float* b_ih3 = (const float*)d_in[19];
    const float* b_hh3 = (const float*)d_in[20];

    // workspace carve-out (256B aligned)
    char* ws = (char*)d_ws;
    size_t off = 0;
    auto carve = [&](size_t elems) -> bf16_t* {
        off = (off + 255) & ~(size_t)255;
        bf16_t* p = (bf16_t*)(ws + off);
        off += elems * sizeof(bf16_t);
        return p;
    };
    const size_t SZ_BK  = (size_t)BROWS * KPSZ;
    const size_t SZ_BH  = (size_t)BROWS * HID;
    const size_t SZ_WE  = (size_t)KPSZ * KPSZ;
    const size_t SZ_WI1 = (size_t)4 * HID * KPSZ;
    const size_t SZ_WH  = (size_t)4 * HID * HID;

    bf16_t* kps_bf   = carve(SZ_BK);
    bf16_t* frame_bf = carve(SZ_BK);
    bf16_t* h0_bf    = carve(SZ_BH);
    bf16_t* h1_bf    = carve(SZ_BH);
    bf16_t* h2_bf    = carve(SZ_BH);
    bf16_t* wemb_bf  = carve(SZ_WE);
    bf16_t* wih1_bf  = carve(SZ_WI1);
    bf16_t* whh1_bf  = carve(SZ_WH);
    bf16_t* wih2_bf  = carve(SZ_WH);
    bf16_t* whh2_bf  = carve(SZ_WH);
    bf16_t* wih3_bf  = carve(SZ_WH);
    bf16_t* whh3_bf  = carve(SZ_WH);

    auto cvt = [&](const float* in, bf16_t* out, size_t n) {
        int n4 = (int)(n / 4);
        int blocks = (n4 + 255) / 256;
        if (blocks > 4096) blocks = 4096;
        cvt_f32_to_bf16<<<blocks, 256, 0, stream>>>(
            (const float4*)in, (v4bf*)out, n4);
    };
    cvt(kps,   kps_bf,  SZ_BK);
    cvt(h0,    h0_bf,   SZ_BH);
    cvt(h1,    h1_bf,   SZ_BH);
    cvt(h2,    h2_bf,   SZ_BH);
    cvt(W_emb, wemb_bf, SZ_WE);
    cvt(w_ih1, wih1_bf, SZ_WI1);
    cvt(w_hh1, whh1_bf, SZ_WH);
    cvt(w_ih2, wih2_bf, SZ_WH);
    cvt(w_hh2, whh2_bf, SZ_WH);
    cvt(w_ih3, wih3_bf, SZ_WH);
    cvt(w_hh3, whh3_bf, SZ_WH);

    // output slots: (h2n, h0n, h1n, h2n, c0n, c1n, c2n), each [B, H] fp32
    float* o = (float*)d_out;
    const size_t S = SZ_BH;
    float* o_h2n_a = o + 0 * S;
    float* o_h0n   = o + 1 * S;
    float* o_h1n   = o + 2 * S;
    float* o_h2n_b = o + 3 * S;
    float* o_c0n   = o + 4 * S;
    float* o_c1n   = o + 5 * S;
    float* o_c2n   = o + 6 * S;

    emb_kernel<<<dim3(KPSZ / BN, BROWS / BM), 256, 0, stream>>>(
        kps_bf, wemb_bf, b_emb, frame_bf);

    dim3 grid(HID / BN, BROWS / BM);
    lstm_cell_kernel<KPSZ><<<grid, 256, 0, stream>>>(
        frame_bf, h0_bf, wih1_bf, whh1_bf, b_ih1, b_hh1, c0,
        o_h0n, nullptr, o_c0n);
    lstm_cell_kernel<HID><<<grid, 256, 0, stream>>>(
        h0_bf, h1_bf, wih2_bf, whh2_bf, b_ih2, b_hh2, c1,
        o_h1n, nullptr, o_c1n);
    lstm_cell_kernel<HID><<<grid, 256, 0, stream>>>(
        h1_bf, h2_bf, wih3_bf, whh3_bf, b_ih3, b_hh3, c2,
        o_h2n_a, o_h2n_b, o_c2n);
}
